// TFA_67070209294733
// MI455X (gfx1250) — compile-verified, matrix operand order checked by reading, not run on hardware
//
#include <hip/hip_runtime.h>
#include <math.h>

typedef __attribute__((ext_vector_type(16))) _Float16 v16h;
typedef __attribute__((ext_vector_type(8)))  float    v8f;

namespace cfg {
constexpr int NB    = 128;   // batch
constexpr int SEQ   = 64;    // frames
constexpr int CH    = 256;   // channels per part
constexpr int PARTS = 16;
constexpr int HID   = 64;
constexpr int XT_ROWS   = SEQ + 4;   // +/-2 zero pad rows
constexpr int XT_STRIDE = CH + 8;    // 264 halfs: bank-conflict padding
constexpr int HT_STRIDE = HID + 8;   // 72 halfs
}

// ---------------------------------------------------------------------------
// 16x16x32 f16 fragment helpers.
// A-matrix (16x32, MxK), per ISA 7.12.2:
//   lane m = lane&15, hs = lane>>4
//   half slot t=2v+e  ->  k = (v>>2)*16 + hs*8 + (v&3)*2 + e
// B fragment is built as the A-fragment of B^T (B loaded column-major).
// ---------------------------------------------------------------------------
__device__ __forceinline__ v16h load_a_frag_g(const float* base, int rstride,
                                              int kstride, int lane) {
  const int m  = lane & 15;
  const int hs = (lane >> 4) & 1;
  const float* p = base + (size_t)m * rstride;
  v16h a;
#pragma unroll
  for (int t = 0; t < 16; ++t) {
    const int v = t >> 1, e = t & 1;
    const int k = ((v >> 2) << 4) + (hs << 3) + ((v & 3) << 1) + e;
    a[t] = (_Float16)p[k * kstride];
  }
  return a;
}

__device__ __forceinline__ v16h load_b_frag_lds(const _Float16* lds, int row_base,
                                                int row_stride, int col_base,
                                                int lane) {
  const int n  = lane & 15;           // output column of this lane
  const int hs = (lane >> 4) & 1;
  const _Float16* p = lds + (size_t)(row_base + n) * row_stride + col_base;
  v16h b;
#pragma unroll
  for (int t = 0; t < 16; ++t) {
    const int v = t >> 1, e = t & 1;
    const int k = ((v >> 2) << 4) + (hs << 3) + ((v & 3) << 1) + e;
    b[t] = p[k];
  }
  return b;
}

__device__ __forceinline__ v8f wmma_f16(v16h a, v16h b, v8f c) {
  return __builtin_amdgcn_wmma_f32_16x16x32_f16(false, a, false, b,
                                                (short)0, c, false, false);
}

__device__ __forceinline__ float sigmoidf_(float z) {
  return 1.0f / (1.0f + __expf(-z));
}

// ---------------------------------------------------------------------------
// One block per (part, batch). 256 threads = 8 waves (wave32).
// ---------------------------------------------------------------------------
__global__ __launch_bounds__(256)
void mtb_gate_kernel(const float* __restrict__ x,
                     const float* __restrict__ W1a,
                     const float* __restrict__ W1b,
                     const float* __restrict__ W2a,
                     const float* __restrict__ W2b,
                     float* __restrict__ out) {
  using namespace cfg;
  __shared__ _Float16 sXt[XT_ROWS * XT_STRIDE];     // x tile, [l+2][c], f16
  __shared__ _Float16 sHt[2][XT_ROWS * HT_STRIDE];  // hidden, [l+2][h], f16

  const int part = blockIdx.x;
  const int b    = blockIdx.y;
  const int tid  = threadIdx.x;
  const int lane = tid & 31;
  const int wave = tid >> 5;
  const int hs   = (lane >> 4) & 1;

  // ---- Stage 0: zero pads + load x tile (f32 -> f16, transposed) ----------
  {
    // zero pad rows of Xt (rows 0,1,SEQ+2,SEQ+3), one column per thread
    sXt[0 * XT_STRIDE + tid]         = (_Float16)0.f;
    sXt[1 * XT_STRIDE + tid]         = (_Float16)0.f;
    sXt[(SEQ + 2) * XT_STRIDE + tid] = (_Float16)0.f;
    sXt[(SEQ + 3) * XT_STRIDE + tid] = (_Float16)0.f;
    if (tid < HID) {
#pragma unroll
      for (int r = 0; r < 2; ++r) {
        sHt[0][r * HT_STRIDE + tid] = (_Float16)0.f;
        sHt[1][r * HT_STRIDE + tid] = (_Float16)0.f;
        sHt[0][(SEQ + 2 + r) * HT_STRIDE + tid] = (_Float16)0.f;
        sHt[1][(SEQ + 2 + r) * HT_STRIDE + tid] = (_Float16)0.f;
      }
    }
    // x[b][l][c][part] ; thread tid owns channel c = tid, loops l
    const int c = tid;
    const float* xp = x + (((size_t)b * SEQ) * CH + c) * PARTS + part;
#pragma unroll 4
    for (int l = 0; l < SEQ; ++l) {
      float v = xp[(size_t)l * CH * PARTS];
      sXt[(2 + l) * XT_STRIDE + c] = (_Float16)v;
    }
  }
  __syncthreads();

  // ---- Stage A: hidden = LeakyReLU(conv3(x)) for both branches ------------
  // wave -> branch br = wave>>2 (0: W1a->H1, 1: W2a->H2), m-tile mt = wave&3
  {
    const int br = wave >> 2;
    const int mt = wave & 3;
    const int m0 = mt * 16;
    const float* Wa = (br ? W2a : W1a) +
                      ((size_t)(part * HID + m0)) * CH * 3;  // [h][c][k]
    v8f acc[4] = {};
    for (int s = 0; s < 3; ++s) {
      for (int kb = 0; kb < CH / 32; ++kb) {
        const int c0 = kb * 32;
        v16h a = load_a_frag_g(Wa + c0 * 3 + s, CH * 3, 3, lane);
#pragma unroll
        for (int nt = 0; nt < 4; ++nt) {
          v16h bf = load_b_frag_lds(sXt, 1 + s + nt * 16, XT_STRIDE, c0, lane);
          acc[nt] = wmma_f16(a, bf, acc[nt]);
        }
      }
    }
    // LeakyReLU + store transposed: Ht[l+2][h]
#pragma unroll
    for (int nt = 0; nt < 4; ++nt) {
#pragma unroll
      for (int v = 0; v < 8; ++v) {
        float f = acc[nt][v];
        f = f > 0.f ? f : 0.01f * f;
        const int row = 2 + nt * 16 + (lane & 15);
        const int col = m0 + v + (hs << 3);
        sHt[br][row * HT_STRIDE + col] = (_Float16)f;
      }
    }
  }
  __syncthreads();

  // ---- Stage B: logits, gates, pools, temporal max ------------------------
  // wave owns channels [wave*32, wave*32+32)
  for (int j = 0; j < 2; ++j) {
    const int m0 = wave * 32 + j * 16;  // channel-tile base
    // A fragments: W1b [c][h] (k=1), W2b [c][h][k] (k=3)
    const float* w1bp = W1b + ((size_t)(part * CH + m0)) * HID;
    const float* w2bp = W2b + ((size_t)(part * CH + m0)) * HID * 3;
    v16h a1[2], a2[3][2];
#pragma unroll
    for (int kb = 0; kb < 2; ++kb)
      a1[kb] = load_a_frag_g(w1bp + kb * 32, HID, 1, lane);
#pragma unroll
    for (int s = 0; s < 3; ++s)
#pragma unroll
      for (int kb = 0; kb < 2; ++kb)
        a2[s][kb] = load_a_frag_g(w2bp + kb * 32 * 3 + s, HID * 3, 3, lane);

    float maxv[8];
#pragma unroll
    for (int v = 0; v < 8; ++v) maxv[v] = -INFINITY;

    for (int nt = 0; nt < 4; ++nt) {
      v8f acc1 = {}, acc2 = {};
#pragma unroll
      for (int kb = 0; kb < 2; ++kb) {
        v16h bf = load_b_frag_lds(sHt[0], 2 + nt * 16, HT_STRIDE, kb * 32, lane);
        acc1 = wmma_f16(a1[kb], bf, acc1);
      }
      for (int s = 0; s < 3; ++s) {
#pragma unroll
        for (int kb = 0; kb < 2; ++kb) {
          v16h bf = load_b_frag_lds(sHt[1], 1 + s + nt * 16, HT_STRIDE,
                                    kb * 32, lane);
          acc2 = wmma_f16(a2[s][kb], bf, acc2);
        }
      }
#pragma unroll
      for (int v = 0; v < 8; ++v) {
        const int c = m0 + v + (hs << 3);
        const int l = nt * 16 + (lane & 15);
        float xs[5];
#pragma unroll
        for (int d = 0; d < 5; ++d)   // rows l..l+4 == padded (l-2..l+2)
          xs[d] = (float)sXt[(l + d) * XT_STRIDE + c];
        const float avg3 = (xs[1] + xs[2] + xs[3]) * (1.0f / 3.0f);
        const float avg5 = (xs[0] + xs[1] + xs[2] + xs[3] + xs[4]) * 0.2f;
        float m3 = xs[2];
        m3 = fmaxf(m3, (l >= 1)       ? xs[1] : -INFINITY);
        m3 = fmaxf(m3, (l <= cfg::SEQ - 2) ? xs[3] : -INFINITY);
        float m5 = m3;
        m5 = fmaxf(m5, (l >= 2)       ? xs[0] : -INFINITY);
        m5 = fmaxf(m5, (l <= cfg::SEQ - 3) ? xs[4] : -INFINITY);
        const float g1 = sigmoidf_(acc1[v]);
        const float g2 = sigmoidf_(acc2[v]);
        const float mval = (avg3 + m3) * g1 + (avg5 + m5) * g2;
        maxv[v] = fmaxf(maxv[v], mval);
      }
    }
    // reduce max over the 16 lanes sharing a channel, then store
#pragma unroll
    for (int v = 0; v < 8; ++v) {
#pragma unroll
      for (int off = 8; off >= 1; off >>= 1)
        maxv[v] = fmaxf(maxv[v], __shfl_xor(maxv[v], off, 16));
      if ((lane & 15) == 0) {
        const int c = m0 + v + (hs << 3);
        out[((size_t)b * cfg::PARTS + part) * cfg::CH + c] = maxv[v];
      }
    }
  }
}

extern "C" void kernel_launch(void* const* d_in, const int* in_sizes, int n_in,
                              void* d_out, int out_size, void* d_ws, size_t ws_size,
                              hipStream_t stream) {
  (void)in_sizes; (void)n_in; (void)d_ws; (void)ws_size; (void)out_size;
  const float* x   = (const float*)d_in[0];
  const float* W1a = (const float*)d_in[1];
  const float* W1b = (const float*)d_in[2];
  const float* W2a = (const float*)d_in[3];
  const float* W2b = (const float*)d_in[4];
  float* out = (float*)d_out;

  dim3 grid(cfg::PARTS, cfg::NB);  // (part, batch)
  dim3 block(256);
  mtb_gate_kernel<<<grid, block, 0, stream>>>(x, W1a, W1b, W2a, W2b, out);
}
